// SVDHead_395136991552
// MI455X (gfx1250) — compile-verified, matrix operand order checked by reading, not run on hardware
//
#include <hip/hip_runtime.h>
#include <hip/hip_bf16.h>
#include <math.h>

typedef __attribute__((ext_vector_type(16))) __bf16 v16bf;
typedef __attribute__((ext_vector_type(8)))  float  v8f;

#define Bc 8
#define Dc 128
#define Nc 3072
#define TILES_PER_B 48   /* N / 64 queries per workgroup */
#define KT 192           /* N / 16 key tiles */
#define FRAG_OFF_FLOATS 8192                       /* partials region: 32KB */
#define FRAG_ELEMS ((size_t)Bc * KT * 4 * 32 * 16) /* bf16 elems per tensor */

// exp(s/sqrt(128)) == exp2(s * (log2(e)/sqrt(128))) : fold both constants
#define KEXP2 (0.08838834764831845f * 1.44269504088896340736f)

__device__ __forceinline__ unsigned short f2bf(float f) {
    unsigned int u = __float_as_uint(f);
    unsigned int r = u + 0x7fffu + ((u >> 16) & 1u);   // round-to-nearest-even
    return (unsigned short)(r >> 16);
}

// Element i of a v16bf fragment maps to K index per ISA 16-bit A/B layout:
// lanes 0-15 hold K {0..7, 16..23}, lanes 16-31 hold K {8..15, 24..31}.
__device__ __forceinline__ int kmap(int i, int lane) {
    return (i < 8 ? i : i + 8) + ((lane >> 4) << 3);
}

union BF16Frag { v16bf v; unsigned short u[16]; };

// ---------------------------------------------------------------------------
// Pre-pass: convert an f32 [B,128,3072] embedding into bf16 WMMA fragments:
// dst[((b*192 + t16)*4 + chunk)*32 + lane][16]  (32B per lane, coalesced).
// Same mapping serves A (lane = row) and B (lane = column) fragments.
// ---------------------------------------------------------------------------
__global__ __launch_bounds__(128) void convert_frags_kernel(
    const float* __restrict__ emb, unsigned short* __restrict__ dst)
{
    const int b  = blockIdx.y;
    const int t16 = blockIdx.x;       // 0..191
    const int tid = threadIdx.x;      // 0..127
    const int c  = tid >> 5;
    const int Lf = tid & 31;
    const int m  = t16 * 16 + (Lf & 15);
    const float* e = emb + (size_t)b * Dc * Nc;
    unsigned short* o = dst + ((((size_t)b * KT + t16) * 4 + c) * 32 + Lf) * 16;
    BF16Frag fr;
    #pragma unroll
    for (int i = 0; i < 16; ++i) {
        int d = c * 32 + kmap(i, Lf);
        fr.u[i] = f2bf(e[(size_t)d * Nc + m]);
    }
    *(v16bf*)o = fr.v;
}

// ---------------------------------------------------------------------------
// Main fused attention (pre-converted fragments): no LDS / barriers in hot
// loop; ping-pong double buffering (no register copies).
// ---------------------------------------------------------------------------
__global__ __launch_bounds__(128) void svd_attn_pre(
    const unsigned short* __restrict__ preA,
    const unsigned short* __restrict__ preB,
    const float* __restrict__ src,
    const float* __restrict__ tgt,
    float* __restrict__ ws)
{
    const int b     = blockIdx.y;
    const int tileQ = blockIdx.x;          // 0..47
    const int tid   = threadIdx.x;
    const int wave  = tid >> 5;
    const int lane  = tid & 31;

    __shared__ float ldsCorr[3 * 64];

    const float* tgtb = tgt + (size_t)b * 3 * Nc;
    const float* srcb = src + (size_t)b * 3 * Nc;

    // A fragments for this wave's 16 query rows (frag tile = tileQ*4 + wave)
    const int fiA = tileQ * 4 + wave;
    v16bf af[4];
    #pragma unroll
    for (int c = 0; c < 4; ++c)
        af[c] = *(const v16bf*)(preA + ((((size_t)b * KT + fiA) * 4 + c) * 32 + lane) * 16);

    float lsum[8], acc0[8], acc1[8], acc2[8];
    #pragma unroll
    for (int i = 0; i < 8; ++i) { lsum[i] = 0.f; acc0[i] = 0.f; acc1[i] = 0.f; acc2[i] = 0.f; }

    const int mLane = lane & 15;
    const unsigned short* preBb = preB + (size_t)b * KT * 4 * 32 * 16;

    v16bf b0[4], b1[4];
    #pragma unroll
    for (int c = 0; c < 4; ++c) {
        b0[c] = *(const v16bf*)(preBb + (((size_t)0 * 4 + c) * 32 + lane) * 16);
        b1[c] = *(const v16bf*)(preBb + (((size_t)1 * 4 + c) * 32 + lane) * 16);
    }

    for (int kt = 0; kt < KT; kt += 2) {
        // ---- tile kt (uses b0), load kt+2 into b0
        {
            const int m0 = kt * 16;
            const float t0 = tgtb[               m0 + mLane];
            const float t1 = tgtb[(size_t)Nc     + m0 + mLane];
            const float t2 = tgtb[(size_t)Nc * 2 + m0 + mLane];
            v8f accS = {0.f, 0.f, 0.f, 0.f, 0.f, 0.f, 0.f, 0.f};
            #pragma unroll
            for (int c = 0; c < 4; ++c)
                accS = __builtin_amdgcn_wmma_f32_16x16x32_bf16(
                    false, af[c], false, b0[c], (short)0, accS, false, false);
            if (kt + 2 < KT) {
                #pragma unroll
                for (int c = 0; c < 4; ++c)
                    b0[c] = *(const v16bf*)(preBb + ((((size_t)kt + 2) * 4 + c) * 32 + lane) * 16);
                __builtin_prefetch(preBb + (size_t)(kt + 3) * 4 * 32 * 16 + lane * 64, 0, 3);
            }
            #pragma unroll
            for (int i = 0; i < 8; ++i) {
                float p = exp2f(accS[i] * KEXP2);
                lsum[i] += p;
                acc0[i] += p * t0;
                acc1[i] += p * t1;
                acc2[i] += p * t2;
            }
        }
        // ---- tile kt+1 (uses b1), load kt+3 into b1
        {
            const int m0 = (kt + 1) * 16;
            const float t0 = tgtb[               m0 + mLane];
            const float t1 = tgtb[(size_t)Nc     + m0 + mLane];
            const float t2 = tgtb[(size_t)Nc * 2 + m0 + mLane];
            v8f accS = {0.f, 0.f, 0.f, 0.f, 0.f, 0.f, 0.f, 0.f};
            #pragma unroll
            for (int c = 0; c < 4; ++c)
                accS = __builtin_amdgcn_wmma_f32_16x16x32_bf16(
                    false, af[c], false, b1[c], (short)0, accS, false, false);
            if (kt + 3 < KT) {
                #pragma unroll
                for (int c = 0; c < 4; ++c)
                    b1[c] = *(const v16bf*)(preBb + ((((size_t)kt + 3) * 4 + c) * 32 + lane) * 16);
            }
            #pragma unroll
            for (int i = 0; i < 8; ++i) {
                float p = exp2f(accS[i] * KEXP2);
                lsum[i] += p;
                acc0[i] += p * t0;
                acc1[i] += p * t1;
                acc2[i] += p * t2;
            }
        }
    }

    // reduce across the 16 lanes (columns) within each half-wave
    #pragma unroll
    for (int i = 0; i < 8; ++i) {
        #pragma unroll
        for (int m = 1; m < 16; m <<= 1) {
            lsum[i] += __shfl_xor(lsum[i], m, 32);
            acc0[i] += __shfl_xor(acc0[i], m, 32);
            acc1[i] += __shfl_xor(acc1[i], m, 32);
            acc2[i] += __shfl_xor(acc2[i], m, 32);
        }
    }

    if ((lane & 15) == 0) {
        int half = lane >> 4;
        #pragma unroll
        for (int i = 0; i < 8; ++i) {
            int r = wave * 16 + half * 8 + i;
            float inv = 1.0f / lsum[i];
            ldsCorr[0 * 64 + r] = acc0[i] * inv;
            ldsCorr[1 * 64 + r] = acc1[i] * inv;
            ldsCorr[2 * 64 + r] = acc2[i] * inv;
        }
    }
    __syncthreads();

    if (tid < 15) {
        float sum = 0.f;
        for (int r = 0; r < 64; ++r) {
            int n = tileQ * 64 + r;
            float v;
            if (tid < 9) {
                int i = tid / 3, j = tid % 3;
                v = srcb[(size_t)i * Nc + n] * ldsCorr[j * 64 + r];
            } else if (tid < 12) {
                v = srcb[(size_t)(tid - 9) * Nc + n];
            } else {
                v = ldsCorr[(tid - 12) * 64 + r];
            }
            sum += v;
        }
        ws[((size_t)b * TILES_PER_B + tileQ) * 16 + tid] = sum;
    }
}

// ---------------------------------------------------------------------------
// Fallback (self-staging through LDS) when d_ws is too small for fragments
// ---------------------------------------------------------------------------
__global__ __launch_bounds__(128) void svd_attn_fallback(
    const float* __restrict__ src_emb,
    const float* __restrict__ tgt_emb,
    const float* __restrict__ src,
    const float* __restrict__ tgt,
    float* __restrict__ ws)
{
    const int b     = blockIdx.y;
    const int tileQ = blockIdx.x;
    const int tid   = threadIdx.x;
    const int wave  = tid >> 5;
    const int lane  = tid & 31;

    __shared__ unsigned short ldsB[4 * 32 * 16];
    __shared__ float ldsTgt[3 * 16];
    __shared__ float ldsCorr[3 * 64];

    const float* seb  = src_emb + (size_t)b * Dc * Nc;
    const float* teb  = tgt_emb + (size_t)b * Dc * Nc;
    const float* tgtb = tgt     + (size_t)b * 3  * Nc;
    const float* srcb = src     + (size_t)b * 3  * Nc;

    const int n0w  = tileQ * 64 + wave * 16;
    const int rowA = n0w + (lane & 15);
    BF16Frag af[4];
    #pragma unroll
    for (int c = 0; c < 4; ++c) {
        #pragma unroll
        for (int i = 0; i < 16; ++i) {
            int d = c * 32 + kmap(i, lane);
            af[c].u[i] = f2bf(seb[(size_t)d * Nc + rowA]);
        }
    }

    float lsum[8], acc0[8], acc1[8], acc2[8];
    #pragma unroll
    for (int i = 0; i < 8; ++i) { lsum[i] = 0.f; acc0[i] = 0.f; acc1[i] = 0.f; acc2[i] = 0.f; }

    const int cS  = tid >> 5;
    const int LfS = tid & 31;
    const int mS  = LfS & 15;

    for (int kt = 0; kt < KT; ++kt) {
        const int m0 = kt * 16;
        #pragma unroll
        for (int i = 0; i < 16; ++i) {
            int d = cS * 32 + kmap(i, LfS);
            ldsB[(cS * 32 + LfS) * 16 + i] = f2bf(teb[(size_t)d * Nc + m0 + mS]);
        }
        if (tid < 48) {
            int d = tid >> 4, j = tid & 15;
            ldsTgt[d * 16 + j] = tgtb[(size_t)d * Nc + m0 + j];
        }
        __syncthreads();

        v8f accS = {0.f, 0.f, 0.f, 0.f, 0.f, 0.f, 0.f, 0.f};
        #pragma unroll
        for (int c = 0; c < 4; ++c) {
            v16bf bvv = *(const v16bf*)&ldsB[(c * 32 + lane) * 16];
            accS = __builtin_amdgcn_wmma_f32_16x16x32_bf16(
                false, af[c].v, false, bvv, (short)0, accS, false, false);
        }

        const float t0 = ldsTgt[      (lane & 15)];
        const float t1 = ldsTgt[16 +  (lane & 15)];
        const float t2 = ldsTgt[32 +  (lane & 15)];
        #pragma unroll
        for (int i = 0; i < 8; ++i) {
            float p = exp2f(accS[i] * KEXP2);
            lsum[i] += p;
            acc0[i] += p * t0;
            acc1[i] += p * t1;
            acc2[i] += p * t2;
        }
        __syncthreads();
    }

    #pragma unroll
    for (int i = 0; i < 8; ++i) {
        #pragma unroll
        for (int m = 1; m < 16; m <<= 1) {
            lsum[i] += __shfl_xor(lsum[i], m, 32);
            acc0[i] += __shfl_xor(acc0[i], m, 32);
            acc1[i] += __shfl_xor(acc1[i], m, 32);
            acc2[i] += __shfl_xor(acc2[i], m, 32);
        }
    }

    if ((lane & 15) == 0) {
        int half = lane >> 4;
        #pragma unroll
        for (int i = 0; i < 8; ++i) {
            int r = wave * 16 + half * 8 + i;
            float inv = 1.0f / lsum[i];
            ldsCorr[0 * 64 + r] = acc0[i] * inv;
            ldsCorr[1 * 64 + r] = acc1[i] * inv;
            ldsCorr[2 * 64 + r] = acc2[i] * inv;
        }
    }
    __syncthreads();

    if (tid < 15) {
        float sum = 0.f;
        for (int r = 0; r < 64; ++r) {
            int n = tileQ * 64 + r;
            float v;
            if (tid < 9) {
                int i = tid / 3, j = tid % 3;
                v = srcb[(size_t)i * Nc + n] * ldsCorr[j * 64 + r];
            } else if (tid < 12) {
                v = srcb[(size_t)(tid - 9) * Nc + n];
            } else {
                v = ldsCorr[(tid - 12) * 64 + r];
            }
            sum += v;
        }
        ws[((size_t)b * TILES_PER_B + tileQ) * 16 + tid] = sum;
    }
}

// ---------------------------------------------------------------------------
// Per-batch: reduce partials, build H, 3x3 SVD via Jacobi on H^T H,
// Kabsch rotation with reflection fix, translation, euler_zyx.
// ---------------------------------------------------------------------------
__global__ void svd_head_final(const float* __restrict__ ws, float* __restrict__ out)
{
    const int b = blockIdx.x;
    if (threadIdx.x != 0) return;

    float P[15];
    for (int j = 0; j < 15; ++j) P[j] = 0.f;
    for (int tq = 0; tq < TILES_PER_B; ++tq)
        for (int j = 0; j < 15; ++j)
            P[j] += ws[((size_t)b * TILES_PER_B + tq) * 16 + j];

    const float invN = 1.0f / (float)Nc;
    float ms[3], mc[3];
    for (int i = 0; i < 3; ++i) { ms[i] = P[9 + i] * invN; mc[i] = P[12 + i] * invN; }

    float H[3][3];
    for (int i = 0; i < 3; ++i)
        for (int j = 0; j < 3; ++j)
            H[i][j] = P[i * 3 + j] - (float)Nc * ms[i] * mc[j];

    float A[3][3];
    for (int i = 0; i < 3; ++i)
        for (int j = 0; j < 3; ++j) {
            float s = 0.f;
            for (int k = 0; k < 3; ++k) s += H[k][i] * H[k][j];
            A[i][j] = s;
        }

    float V[3][3] = {{1,0,0},{0,1,0},{0,0,1}};
    for (int sweep = 0; sweep < 20; ++sweep) {
        for (int pi = 0; pi < 3; ++pi) {
            const int p = (pi == 2) ? 1 : 0;
            const int q = (pi == 0) ? 1 : 2;
            float apq = A[p][q];
            if (fabsf(apq) < 1e-30f) continue;
            float tau = (A[q][q] - A[p][p]) / (2.0f * apq);
            float tt  = (tau >= 0.f ? 1.f : -1.f) / (fabsf(tau) + sqrtf(1.f + tau * tau));
            float cc  = 1.0f / sqrtf(1.f + tt * tt);
            float ssn = tt * cc;
            float app = A[p][p], aqq = A[q][q];
            A[p][p] = app - tt * apq;
            A[q][q] = aqq + tt * apq;
            A[p][q] = 0.f; A[q][p] = 0.f;
            int k = 3 - p - q;
            float akp = A[k][p], akq = A[k][q];
            A[k][p] = cc * akp - ssn * akq; A[p][k] = A[k][p];
            A[k][q] = ssn * akp + cc * akq; A[q][k] = A[k][q];
            for (int r = 0; r < 3; ++r) {
                float vp = V[r][p], vq = V[r][q];
                V[r][p] = cc * vp - ssn * vq;
                V[r][q] = ssn * vp + cc * vq;
            }
        }
    }

    float ev[3] = {A[0][0], A[1][1], A[2][2]};
    int idx[3] = {0, 1, 2};
    for (int a = 0; a < 2; ++a)
        for (int c2 = a + 1; c2 < 3; ++c2)
            if (ev[idx[c2]] > ev[idx[a]]) { int t = idx[a]; idx[a] = idx[c2]; idx[c2] = t; }

    float Vs[3][3], U[3][3];
    for (int k = 0; k < 3; ++k) {
        int s0 = idx[k];
        float sig = sqrtf(fmaxf(ev[s0], 0.f));
        float inv = 1.0f / fmaxf(sig, 1e-20f);
        for (int r = 0; r < 3; ++r) Vs[r][k] = V[r][s0];
        for (int r = 0; r < 3; ++r) {
            float s = 0.f;
            for (int c3 = 0; c3 < 3; ++c3) s += H[r][c3] * V[c3][s0];
            U[r][k] = s * inv;
        }
    }

    float Rm[3][3];
    for (int i = 0; i < 3; ++i)
        for (int j = 0; j < 3; ++j) {
            float s = 0.f;
            for (int k = 0; k < 3; ++k) s += Vs[i][k] * U[j][k];
            Rm[i][j] = s;
        }
    float det = Rm[0][0]*(Rm[1][1]*Rm[2][2]-Rm[1][2]*Rm[2][1])
              - Rm[0][1]*(Rm[1][0]*Rm[2][2]-Rm[1][2]*Rm[2][0])
              + Rm[0][2]*(Rm[1][0]*Rm[2][1]-Rm[1][1]*Rm[2][0]);
    if (det < 0.f) {
        for (int r = 0; r < 3; ++r) Vs[r][2] = -Vs[r][2];
        for (int i = 0; i < 3; ++i)
            for (int j = 0; j < 3; ++j) {
                float s = 0.f;
                for (int k = 0; k < 3; ++k) s += Vs[i][k] * U[j][k];
                Rm[i][j] = s;
            }
    }

    float tv[3];
    for (int i = 0; i < 3; ++i) {
        float s = 0.f;
        for (int k = 0; k < 3; ++k) s += Rm[i][k] * ms[k];
        tv[i] = -s + mc[i];
    }

    float sy = sqrtf(Rm[0][0]*Rm[0][0] + Rm[1][0]*Rm[1][0]);
    float ex = atan2f(Rm[2][1], Rm[2][2]);
    float ey = atan2f(-Rm[2][0], sy);
    float ez = atan2f(Rm[1][0], Rm[0][0]);

    for (int i = 0; i < 3; ++i)
        for (int j = 0; j < 3; ++j)
            out[b * 9 + i * 3 + j] = Rm[i][j];
    out[72 + b * 3 + 0] = tv[0];
    out[72 + b * 3 + 1] = tv[1];
    out[72 + b * 3 + 2] = tv[2];
    out[96 + b * 3 + 0] = ex;
    out[96 + b * 3 + 1] = ey;
    out[96 + b * 3 + 2] = ez;
}

extern "C" void kernel_launch(void* const* d_in, const int* in_sizes, int n_in,
                              void* d_out, int out_size, void* d_ws, size_t ws_size,
                              hipStream_t stream) {
    const float* src_emb = (const float*)d_in[0];
    const float* tgt_emb = (const float*)d_in[1];
    const float* src     = (const float*)d_in[2];
    const float* tgt     = (const float*)d_in[3];
    float* out = (float*)d_out;
    float* ws  = (float*)d_ws;

    const size_t frag_bytes = FRAG_ELEMS * sizeof(unsigned short);   // 6.29 MB each
    const size_t need = (size_t)FRAG_OFF_FLOATS * 4 + 2 * frag_bytes;

    dim3 grid(TILES_PER_B, Bc);
    if (ws_size >= need) {
        unsigned short* preA = (unsigned short*)((char*)d_ws + FRAG_OFF_FLOATS * 4);
        unsigned short* preB = (unsigned short*)((char*)preA + frag_bytes);
        dim3 cgrid(KT, Bc);
        convert_frags_kernel<<<cgrid, 128, 0, stream>>>(src_emb, preA);
        convert_frags_kernel<<<cgrid, 128, 0, stream>>>(tgt_emb, preB);
        svd_attn_pre<<<grid, 128, 0, stream>>>(preA, preB, src, tgt, ws);
    } else {
        svd_attn_fallback<<<grid, 128, 0, stream>>>(src_emb, tgt_emb, src, tgt, ws);
    }
    svd_head_final<<<dim3(Bc), 32, 0, stream>>>(ws, out);
}
